// SlotAttention_257698038296
// MI455X (gfx1250) — matrix operand order repaired and checked by reading, not compiled, not run on hardware
//
#include <hip/hip_runtime.h>
#include <hip/hip_bf16.h>
#include <math.h>

typedef __attribute__((ext_vector_type(16))) _Float16 v16h;
typedef __attribute__((ext_vector_type(8)))  _Float16 v8h;
typedef __attribute__((ext_vector_type(8)))  float    v8f;

#define DIM   256
#define NTOK  4096
#define BATCH 64
#define NSLOT 8
#define HID   256
#define N_ITERS 3
#define LN_EPS 1e-5f
#define ATTN_EPS 1e-8f
#define KSPLIT 4                       // K-split for updates GEMM

// ---------------- WMMA fragment helpers (CDNA5 wave32 layouts) ----------------

// A matrix 16x32 f16: lane l holds row m = l%16; lanes 0-15: K = kk+[0..15];
// lanes 16-31: K = kk+16+[0..15].
__device__ __forceinline__ v16h load_fragA(const _Float16* base, int ld, int kk) {
  const int lane = threadIdx.x & 31;
  const int m    = lane & 15;
  const int kb   = (lane >> 4) * 16;
  const _Float16* p = base + (size_t)m * ld + kk + kb;
  v8h lo = *(const v8h*)(p);
  v8h hi = *(const v8h*)(p + 8);
  return __builtin_shufflevector(lo, hi, 0,1,2,3,4,5,6,7,8,9,10,11,12,13,14,15);
}

// B matrix 32x16 f16, supplied as "Bt": row n of Bt = column n of B (contiguous K).
// lane l holds column n = l%16; lanes 0-15: K = kk+[0..15]; lanes 16-31: K = kk+16+[0..15].
__device__ __forceinline__ v16h load_fragB(const _Float16* baseT, int ld, int kk) {
  const int lane = threadIdx.x & 31;
  const int n    = lane & 15;
  const int kb   = (lane >> 4) * 16;
  const _Float16* p = baseT + (size_t)n * ld + kk + kb;
  v8h lo = *(const v8h*)(p);
  v8h hi = *(const v8h*)(p + 8);
  return __builtin_shufflevector(lo, hi, 0,1,2,3,4,5,6,7,8,9,10,11,12,13,14,15);
}

__device__ __forceinline__ v8f wmma_f32_f16(v16h a, v16h b, v8f c) {
  return __builtin_amdgcn_wmma_f32_16x16x32_f16(false, a, false, b, (short)0, c, false, false);
}

// ---------------- small prep kernels ----------------

__global__ void init_slots_kernel(const float* __restrict__ noise,
                                  const float* __restrict__ mu,
                                  const float* __restrict__ sigma,
                                  float* __restrict__ slots) {
  int i = blockIdx.x * 256 + threadIdx.x;             // B*S*D = 131072
  int j = i % (NSLOT * DIM);
  slots[i] = mu[j] + sigma[j] * (noise[i] * 0.1f);
}

__global__ void prep_weights_kernel(const float* __restrict__ Wk,
                                    const float* __restrict__ Wv,
                                    _Float16* __restrict__ WkT,
                                    _Float16* __restrict__ WvT) {
  int i = blockIdx.x * 256 + threadIdx.x;             // D*D
  int k = i / DIM, n = i % DIM;
  WkT[n * DIM + k] = (_Float16)Wk[i];
  WvT[n * DIM + k] = (_Float16)Wv[i];
}

__global__ void zero_pads_kernel(_Float16* __restrict__ qf,
                                 _Float16* __restrict__ attnf) {
  int i = blockIdx.x * 256 + threadIdx.x;             // covers B*8*NTOK
  if (i < BATCH * 8 * DIM) {
    int b = i / (8 * DIM), r = i % (8 * DIM);
    qf[((size_t)b * 16 + 8) * DIM + r] = (_Float16)0.f;
  }
  int b2 = i / (8 * NTOK), r2 = i % (8 * NTOK);
  attnf[((size_t)b2 * 16 + 8) * NTOK + r2] = (_Float16)0.f;
}

// zero both the per-(b,s) renorm sums and the updates accumulator (for K-split atomics)
__global__ void zero_accum_kernel(float* __restrict__ rowsum, float* __restrict__ upd) {
  int i = blockIdx.x * 256 + threadIdx.x;             // B*NSLOT*DIM = 131072
  upd[i] = 0.f;
  if (i < BATCH * NSLOT) rowsum[i] = 0.f;
}

// ---------------- fused LN + k/v projection (WMMA) ----------------
// grid(x = B*N/16 row tiles, y = 0:k / 1:v), 256 threads = 8 waves.
__global__ void __launch_bounds__(256) gemm_kv_kernel(
    const float* __restrict__ inputs,
    const float* __restrict__ lng, const float* __restrict__ lnb,
    const _Float16* __restrict__ WkT, const _Float16* __restrict__ WvT,
    const float* __restrict__ bk, const float* __restrict__ bv,
    _Float16* __restrict__ k_out, _Float16* __restrict__ vT_out) {
  __shared__ __align__(16) _Float16 xs[16 * DIM];
  const int r0   = blockIdx.x * 16;
  const int wave = threadIdx.x >> 5;
  const int lane = threadIdx.x & 31;

  // LayerNorm 2 rows per wave into LDS f16 tile
  for (int rr = 0; rr < 2; ++rr) {
    const int row = 2 * wave + rr;
    const float* xr = inputs + (size_t)(r0 + row) * DIM;
    float v[8]; float s = 0.f;
    #pragma unroll
    for (int i = 0; i < 8; ++i) { v[i] = xr[lane + 32 * i]; s += v[i]; }
    #pragma unroll
    for (int off = 16; off >= 1; off >>= 1) s += __shfl_xor(s, off, 32);
    const float mean = s * (1.0f / DIM);
    float s2 = 0.f;
    #pragma unroll
    for (int i = 0; i < 8; ++i) { float d = v[i] - mean; s2 += d * d; }
    #pragma unroll
    for (int off = 16; off >= 1; off >>= 1) s2 += __shfl_xor(s2, off, 32);
    const float rstd = rsqrtf(s2 * (1.0f / DIM) + LN_EPS);
    #pragma unroll
    for (int i = 0; i < 8; ++i) {
      const int c = lane + 32 * i;
      xs[row * DIM + c] = (_Float16)(((v[i] - mean) * rstd) * lng[c] + lnb[c]);
    }
  }
  __syncthreads();

  const bool isV = (blockIdx.y == 1);
  const _Float16* Wt   = isV ? WvT : WkT;
  const float*    bias = isV ? bv  : bk;

  for (int j = 0; j < 2; ++j) {
    const int c0 = (2 * wave + j) * 16;
    v8f acc = {};
    #pragma unroll
    for (int kt = 0; kt < 8; ++kt) {
      v16h a = load_fragA(xs, DIM, kt * 32);           // ds_load from LDS
      v16h b = load_fragB(Wt + (size_t)c0 * DIM, DIM, kt * 32);
      acc = wmma_f32_f16(a, b, acc);
    }
    const int ncol  = c0 + (lane & 15);
    const int mbase = (lane >> 4) * 8;
    const float bval = bias[ncol];
    if (!isV) {
      #pragma unroll
      for (int g = 0; g < 8; ++g) {
        const int row = r0 + mbase + g;
        k_out[(size_t)row * DIM + ncol] = (_Float16)(acc[g] + bval);
      }
    } else {
      #pragma unroll
      for (int g = 0; g < 8; ++g) {
        const int tok  = r0 + mbase + g;
        const int b_   = tok >> 12;                    // /4096
        const int nloc = tok & (NTOK - 1);
        vT_out[((size_t)b_ * DIM + ncol) * NTOK + nloc] = (_Float16)(acc[g] + bval);
      }
    }
  }
}

// ---------------- q = LN(slots) @ Wq + bq (small, block per (b,s)) ----------------
__global__ void __launch_bounds__(256) qproj_kernel(
    const float* __restrict__ slots,
    const float* __restrict__ lng, const float* __restrict__ lnb,
    const float* __restrict__ Wq, const float* __restrict__ bq,
    _Float16* __restrict__ qf) {
  __shared__ float red[DIM];
  __shared__ float lnv[DIM];
  const int bs = blockIdx.x;                           // b*8+s
  const int b = bs >> 3, s = bs & 7;
  const int t = threadIdx.x;
  const float x = slots[(size_t)bs * DIM + t];
  red[t] = x; __syncthreads();
  for (int off = 128; off > 0; off >>= 1) { if (t < off) red[t] += red[t + off]; __syncthreads(); }
  const float mean = red[0] * (1.f / DIM);
  __syncthreads();
  const float d = x - mean;
  red[t] = d * d; __syncthreads();
  for (int off = 128; off > 0; off >>= 1) { if (t < off) red[t] += red[t + off]; __syncthreads(); }
  const float rstd = rsqrtf(red[0] * (1.f / DIM) + LN_EPS);
  __syncthreads();
  lnv[t] = d * rstd * lng[t] + lnb[t];
  __syncthreads();
  float acc = bq[t];
  for (int k = 0; k < DIM; ++k) acc += lnv[k] * Wq[(size_t)k * DIM + t];
  qf[((size_t)b * 16 + s) * DIM + t] = (_Float16)acc;
}

// ---------------- dots = scale * q @ k^T (WMMA) ----------------
// grid = B*32 blocks, 8 waves each; wave -> one 16-token column tile.
__global__ void __launch_bounds__(256) dots_kernel(
    const _Float16* __restrict__ qf, const _Float16* __restrict__ kf,
    float* __restrict__ dots) {
  const int wg   = blockIdx.x * 8 + (threadIdx.x >> 5);
  const int b    = wg >> 8;                            // 256 tiles per batch
  const int n0   = (wg & 255) * 16;
  const int lane = threadIdx.x & 31;
  const _Float16* A  = qf + (size_t)b * 16 * DIM;
  const _Float16* Bt = kf + ((size_t)b * NTOK + n0) * DIM;
  v8f acc = {};
  #pragma unroll
  for (int kt = 0; kt < 8; ++kt) {
    v16h a  = load_fragA(A, DIM, kt * 32);
    v16h bb = load_fragB(Bt, DIM, kt * 32);
    acc = wmma_f32_f16(a, bb, acc);
  }
  const float scale = 0.0625f;                         // 256^-0.5
  const int ncol  = n0 + (lane & 15);
  const int mbase = (lane >> 4) * 8;
  #pragma unroll
  for (int g = 0; g < 8; ++g)
    dots[((size_t)b * 16 + mbase + g) * NTOK + ncol] = acc[g] * scale;
}

// ---------------- softmax over slots + accumulate per-slot row sums ----------------
__global__ void __launch_bounds__(256) softmax_kernel(
    float* __restrict__ dots, float* __restrict__ rowsum) {
  __shared__ float ssum[NSLOT];
  const int b = blockIdx.x;
  const int n = blockIdx.y * 256 + threadIdx.x;
  if (threadIdx.x < NSLOT) ssum[threadIdx.x] = 0.f;
  __syncthreads();
  float v[NSLOT]; float m = -1e30f;
  #pragma unroll
  for (int s = 0; s < NSLOT; ++s) { v[s] = dots[((size_t)b * 16 + s) * NTOK + n]; m = fmaxf(m, v[s]); }
  float tot = 0.f;
  #pragma unroll
  for (int s = 0; s < NSLOT; ++s) { v[s] = __expf(v[s] - m); tot += v[s]; }
  const float inv = 1.f / tot;
  #pragma unroll
  for (int s = 0; s < NSLOT; ++s) {
    const float a = v[s] * inv + ATTN_EPS;
    dots[((size_t)b * 16 + s) * NTOK + n] = a;
    atomicAdd(&ssum[s], a);
  }
  __syncthreads();
  if (threadIdx.x < NSLOT) atomicAdd(&rowsum[b * NSLOT + threadIdx.x], ssum[threadIdx.x]);
}

// ---------------- renorm over inputs; emit f32 attn output + f16 attn ----------------
__global__ void __launch_bounds__(256) normalize_kernel(
    const float* __restrict__ dots, const float* __restrict__ rowsum,
    float* __restrict__ attn_out, _Float16* __restrict__ attnf) {
  const size_t i = (size_t)blockIdx.x * 256 + threadIdx.x;   // B*8*NTOK
  const int n  = (int)(i & (NTOK - 1));
  const int bs = (int)(i >> 12);
  const int b = bs >> 3, s = bs & 7;
  const float a = dots[((size_t)b * 16 + s) * NTOK + n] / rowsum[bs];
  attn_out[i] = a;
  attnf[((size_t)b * 16 + s) * NTOK + n] = (_Float16)a;
}

// ---------------- updates = attn @ v via v^T (WMMA, K-split + atomics) ----------------
// grid(B, 2, KSPLIT); 8 waves; wave -> one 16-wide d column tile over 1/KSPLIT of K.
__global__ void __launch_bounds__(256) updates_kernel(
    const _Float16* __restrict__ attnf, const _Float16* __restrict__ vT,
    float* __restrict__ upd) {
  const int b    = blockIdx.x;
  const int wave = threadIdx.x >> 5;
  const int lane = threadIdx.x & 31;
  const int c0   = (blockIdx.y * 8 + wave) * 16;
  const int k0   = blockIdx.z * (NTOK / KSPLIT);
  const _Float16* A  = attnf + (size_t)b * 16 * NTOK + k0;
  const _Float16* Bt = vT + ((size_t)b * DIM + c0) * NTOK + k0;
  const int steps = NTOK / KSPLIT / 32;                // 32 WMMA steps
  v8f acc = {};
  for (int kt = 0; kt < steps; ++kt) {
    // prefetch the vT stream two K-steps ahead (lowers to global_prefetch_b8)
    if (kt + 2 < steps)
      __builtin_prefetch((const void*)(Bt + (size_t)(lane & 15) * NTOK + (kt + 2) * 32), 0, 0);
    v16h a  = load_fragA(A, NTOK, kt * 32);
    v16h bb = load_fragB(Bt, NTOK, kt * 32);
    acc = wmma_f32_f16(a, bb, acc);
  }
  if (lane < 16) {                                     // rows M = g (< 8 kept)
    const int col = c0 + lane;
    #pragma unroll
    for (int g = 0; g < 8; ++g)
      atomicAdd(&upd[((size_t)b * NSLOT + g) * DIM + col], acc[g]);
  }
}

// ---------------- GRU cell + LN + MLP residual (block per (b,s)) ----------------
__global__ void __launch_bounds__(256) gru_mlp_kernel(
    const float* __restrict__ upd, float* __restrict__ slots,
    const float* __restrict__ w_ih, const float* __restrict__ w_hh,
    const float* __restrict__ b_ih, const float* __restrict__ b_hh,
    const float* __restrict__ lng, const float* __restrict__ lnb,
    const float* __restrict__ w1, const float* __restrict__ b1,
    const float* __restrict__ w2, const float* __restrict__ b2) {
  __shared__ float u[DIM], h[DIM], red[DIM], lnv[DIM], hid[DIM];
  const int bs = blockIdx.x;
  const int t  = threadIdx.x;
  u[t] = upd[(size_t)bs * DIM + t];
  h[t] = slots[(size_t)bs * DIM + t];
  __syncthreads();
  float ir = b_ih[t], iz = b_ih[DIM + t], in_ = b_ih[2 * DIM + t];
  float hr = b_hh[t], hz = b_hh[DIM + t], hn  = b_hh[2 * DIM + t];
  const float* wir = w_ih + (size_t)t * DIM;
  const float* wiz = w_ih + (size_t)(DIM + t) * DIM;
  const float* win = w_ih + (size_t)(2 * DIM + t) * DIM;
  const float* whr = w_hh + (size_t)t * DIM;
  const float* whz = w_hh + (size_t)(DIM + t) * DIM;
  const float* whn = w_hh + (size_t)(2 * DIM + t) * DIM;
  for (int k = 0; k < DIM; ++k) {
    const float uk = u[k], hk = h[k];
    ir += uk * wir[k]; iz += uk * wiz[k]; in_ += uk * win[k];
    hr += hk * whr[k]; hz += hk * whz[k]; hn  += hk * whn[k];
  }
  const float r  = 1.f / (1.f + __expf(-(ir + hr)));
  const float z  = 1.f / (1.f + __expf(-(iz + hz)));
  const float nn = tanhf(in_ + r * hn);
  const float hnew = (1.f - z) * nn + z * h[t];
  red[t] = hnew; __syncthreads();
  for (int off = 128; off > 0; off >>= 1) { if (t < off) red[t] += red[t + off]; __syncthreads(); }
  const float mean = red[0] * (1.f / DIM);
  __syncthreads();
  const float d = hnew - mean;
  red[t] = d * d; __syncthreads();
  for (int off = 128; off > 0; off >>= 1) { if (t < off) red[t] += red[t + off]; __syncthreads(); }
  const float rstd = rsqrtf(red[0] * (1.f / DIM) + LN_EPS);
  __syncthreads();
  lnv[t] = d * rstd * lng[t] + lnb[t];
  __syncthreads();
  float a1 = b1[t];
  for (int k = 0; k < DIM; ++k) a1 += lnv[k] * w1[(size_t)k * HID + t];
  hid[t] = fmaxf(a1, 0.f);
  __syncthreads();
  float o = b2[t];
  for (int k = 0; k < HID; ++k) o += hid[k] * w2[(size_t)k * DIM + t];
  slots[(size_t)bs * DIM + t] = hnew + o;
}

// ---------------- host-side launcher ----------------
extern "C" void kernel_launch(void* const* d_in, const int* in_sizes, int n_in,
                              void* d_out, int out_size, void* d_ws, size_t ws_size,
                              hipStream_t stream) {
  (void)in_sizes; (void)n_in; (void)out_size; (void)ws_size;
  const float* inputs   = (const float*)d_in[0];
  const float* noise    = (const float*)d_in[1];
  const float* slots_mu = (const float*)d_in[2];
  const float* slots_sg = (const float*)d_in[3];
  const float* Wq = (const float*)d_in[4];  const float* bq = (const float*)d_in[5];
  const float* Wk = (const float*)d_in[6];  const float* bk = (const float*)d_in[7];
  const float* Wv = (const float*)d_in[8];  const float* bv = (const float*)d_in[9];
  const float* w_ih = (const float*)d_in[10]; const float* w_hh = (const float*)d_in[11];
  const float* b_ih = (const float*)d_in[12]; const float* b_hh = (const float*)d_in[13];
  const float* mlp_w1 = (const float*)d_in[14]; const float* mlp_b1 = (const float*)d_in[15];
  const float* mlp_w2 = (const float*)d_in[16]; const float* mlp_b2 = (const float*)d_in[17];
  const float* ln_in_g = (const float*)d_in[18]; const float* ln_in_b = (const float*)d_in[19];
  const float* ln_s_g  = (const float*)d_in[20]; const float* ln_s_b  = (const float*)d_in[21];
  const float* ln_ff_g = (const float*)d_in[22]; const float* ln_ff_b = (const float*)d_in[23];

  float* slots_out = (float*)d_out;                               // (B,S,D)
  float* attn_out  = (float*)d_out + (size_t)BATCH * NSLOT * DIM; // (B,S,N)

  char* p = (char*)d_ws;
  auto carve = [&](size_t bytes) -> void* {
    void* r = (void*)p; p += (bytes + 255) & ~(size_t)255; return r;
  };
  _Float16* kf     = (_Float16*)carve((size_t)BATCH * NTOK * DIM * 2);
  _Float16* vT     = (_Float16*)carve((size_t)BATCH * DIM * NTOK * 2);
  _Float16* WkT    = (_Float16*)carve((size_t)DIM * DIM * 2);
  _Float16* WvT    = (_Float16*)carve((size_t)DIM * DIM * 2);
  _Float16* qf     = (_Float16*)carve((size_t)BATCH * 16 * DIM * 2);
  float*    dots   = (float*)carve((size_t)BATCH * 16 * NTOK * 4);
  _Float16* attnf  = (_Float16*)carve((size_t)BATCH * 16 * NTOK * 2);
  float*    rowsum = (float*)carve((size_t)BATCH * NSLOT * 4);
  float*    upd    = (float*)carve((size_t)BATCH * NSLOT * DIM * 4);

  init_slots_kernel<<<BATCH * NSLOT * DIM / 256, 256, 0, stream>>>(noise, slots_mu, slots_sg, slots_out);
  prep_weights_kernel<<<DIM * DIM / 256, 256, 0, stream>>>(Wk, Wv, WkT, WvT);
  zero_pads_kernel<<<BATCH * 8 * NTOK / 256, 256, 0, stream>>>(qf, attnf);
  gemm_kv_kernel<<<dim3(BATCH * NTOK / 16, 2), 256, 0, stream>>>(
      inputs, ln_in_g, ln_in_b, WkT, WvT, bk, bv, kf, vT);

  for (int it = 0; it < N_ITERS; ++it) {
    qproj_kernel<<<BATCH * NSLOT, 256, 0, stream>>>(slots_out, ln_s_g, ln_s_b, Wq, bq, qf);
    dots_kernel<<<BATCH * 32, 256, 0, stream>>>(qf, kf, dots);
    zero_accum_kernel<<<BATCH * NSLOT * DIM / 256, 256, 0, stream>>>(rowsum, upd);
    softmax_kernel<<<dim3(BATCH, NTOK / 256), 256, 0, stream>>>(dots, rowsum);
    normalize_kernel<<<BATCH * NSLOT * NTOK / 256, 256, 0, stream>>>(dots, rowsum, attn_out, attnf);
    updates_kernel<<<dim3(BATCH, 2, KSPLIT), 256, 0, stream>>>(attnf, vT, upd);
    gru_mlp_kernel<<<BATCH * NSLOT, 256, 0, stream>>>(
        upd, slots_out, w_ih, w_hh, b_ih, b_hh,
        ln_ff_g, ln_ff_b, mlp_w1, mlp_b1, mlp_w2, mlp_b2);
  }
}
